// GNNTox_25245817765958
// MI455X (gfx1250) — matrix operand order repaired and checked by reading, not compiled, not run on hardware
//
#include <hip/hip_runtime.h>

typedef __attribute__((ext_vector_type(16))) _Float16 v16h;
typedef __attribute__((ext_vector_type(8)))  _Float16 v8h;
typedef __attribute__((ext_vector_type(8)))  float    v8f;

#define THREADS 256

static __device__ __forceinline__ long long gtid() {
    return (long long)blockIdx.x * blockDim.x + threadIdx.x;
}

// ---------------- utility ----------------
__global__ void zero_f(float* __restrict__ p, long long n) {
    long long t = gtid();
    if (t < n) p[t] = 0.0f;
}

// ---------------- degree / norm ----------------
__global__ void degree_k(const long long* __restrict__ dst, float* __restrict__ deg, long long E) {
    long long t = gtid();
    if (t < E) atomicAdd(&deg[dst[t]], 1.0f);
}

__global__ void invsqrt_k(const float* __restrict__ deg, float* __restrict__ isd, long long N) {
    long long t = gtid();
    if (t < N) isd[t] = rsqrtf(deg[t] + 1.0f);
}

// ---------------- layer 1 dense part: h = x @ W1  (x:[N,1], W1:[1,64]) ----------------
__global__ void layer1_k(const float* __restrict__ x, const float* __restrict__ W1,
                         float* __restrict__ h, long long N) {
    long long t = gtid();
    if (t < N * 64) {
        long long n = t >> 6;
        int f = (int)(t & 63);
        h[t] = x[n] * W1[f];
    }
}

// ---------------- one-shot: Wt[f*K + k] = (f16) W[k*F + f] ----------------
__global__ void wt_f16_k(const float* __restrict__ W, _Float16* __restrict__ Wt, int K, int F) {
    int t = blockIdx.x * blockDim.x + threadIdx.x;
    if (t < K * F) {
        int k = t / F;
        int f = t % F;
        Wt[(size_t)f * K + k] = (_Float16)W[t];
    }
}

// ---------------- WMMA GEMM: out[N,F] = A16[N,K] @ Wt16[F,K]^T (f32 accumulate) ---------
// A16 row-major f16; Wt column-major f16 (Wt[f*K+k]). Requires numTiles*16 rows valid.
template<int K, int F>
__global__ void gemm_wmma_k(const _Float16* __restrict__ A, const _Float16* __restrict__ Wt,
                            float* __restrict__ out, int numTiles) {
    const int lane = (int)(threadIdx.x & 31u);
    const int wave = (int)(threadIdx.x >> 5u);
    const int tile = (int)blockIdx.x * ((int)blockDim.x >> 5) + wave;
    if (tile >= numTiles) return;               // wave-uniform: EXEC all-ones below

    const int m  = lane & 15;                   // A row-in-tile, B/D column-in-tile
    const int hf = lane >> 4;                   // lane half selector

    // Preload all A fragments for this 16-row tile.
    // A-layout per lane: e=0..7 -> K=8*hf+e ; e=8..15 -> K=16+8*hf+(e-8)  (two contiguous runs)
    v16h afrag[K / 32];
    const v8h* Ap = (const v8h*)(A + (size_t)(tile * 16 + m) * K);
#pragma unroll
    for (int kc = 0; kc < K / 32; ++kc) {
        const int k0 = kc * 32;
        v8h lo = Ap[(k0 + hf * 8) >> 3];
        v8h hi = Ap[(k0 + 16 + hf * 8) >> 3];
        afrag[kc] = __builtin_shufflevector(lo, hi,
                        0, 1, 2, 3, 4, 5, 6, 7, 8, 9, 10, 11, 12, 13, 14, 15);
    }

#pragma unroll
    for (int ct = 0; ct < F / 16; ++ct) {
        const int col = ct * 16 + m;
        const v8h* Bp = (const v8h*)(Wt + (size_t)col * K);
        v8f c = {};
#pragma unroll
        for (int kc = 0; kc < K / 32; ++kc) {
            const int k0 = kc * 32;
            // B-layout per lane: e=0..7 -> K = k0 + 8*hf + e ; e=8..15 -> K = k0 + 16 + 8*hf + (e-8)
            v8h b0 = Bp[(k0 + hf * 8) >> 3];
            v8h b1 = Bp[(k0 + 16 + hf * 8) >> 3];
            v16h b = __builtin_shufflevector(b0, b1,
                         0, 1, 2, 3, 4, 5, 6, 7, 8, 9, 10, 11, 12, 13, 14, 15);
            c = __builtin_amdgcn_wmma_f32_16x16x32_f16(
                    false, afrag[kc], false, b, (short)0, c, false, false);
        }
#pragma unroll
        for (int r = 0; r < 8; ++r) {
            // D layout: VGPR r holds M = r + 8*hf, column = lane&15
            out[(size_t)(tile * 16 + r + hf * 8) * F + col] = c[r];
        }
    }
}

// ---------------- scalar tail GEMM for rows >= numTiles*16 (not used when N%16==0) -----
template<int K, int F>
__global__ void gemm_tail_k(const _Float16* __restrict__ A, const _Float16* __restrict__ Wt,
                            float* __restrict__ out, int rowStart, int N) {
    int t = blockIdx.x * blockDim.x + threadIdx.x;
    int row = rowStart + t / F;
    int col = t % F;
    if (row >= N) return;
    float acc = 0.0f;
    for (int k = 0; k < K; ++k)
        acc += (float)A[(size_t)row * K + k] * (float)Wt[(size_t)col * K + k];
    out[(size_t)row * F + col] = acc;
}

// ---------------- edge aggregation: agg[dst] += h[src] * isd[src]*isd[dst] (float4) ----
template<int F>
__global__ void aggregate_k(const float* __restrict__ h, const long long* __restrict__ src,
                            const long long* __restrict__ dst, const float* __restrict__ isd,
                            float* __restrict__ agg, long long E) {
    long long t = gtid();
    if (t >= E * (F / 4)) return;
    long long e = t / (F / 4);
    int f4 = (int)(t % (F / 4));
    long long s = src[e];
    long long d = dst[e];
    float norm = isd[s] * isd[d];
    float4 hv = ((const float4*)(h + s * F))[f4];
    float* ap = agg + d * F + (size_t)f4 * 4;
    atomicAdd(ap + 0, hv.x * norm);
    atomicAdd(ap + 1, hv.y * norm);
    atomicAdd(ap + 2, hv.z * norm);
    atomicAdd(ap + 3, hv.w * norm);
}

// ---------------- self-loop + bias + relu, emit activation in f16 (GEMM input) ---------
template<int F>
__global__ void self_bias_relu_f16_k(const float* __restrict__ agg, const float* __restrict__ h,
                                     const float* __restrict__ isd, const float* __restrict__ bias,
                                     _Float16* __restrict__ act, long long N) {
    long long t = gtid();
    if (t >= N * F) return;
    long long n = t / F;
    int f = (int)(t % F);
    float is = isd[n];
    float v = agg[t] + h[t] * is * is + bias[f];
    act[t] = (_Float16)fmaxf(v, 0.0f);
}

// ---------------- self-loop + bias, f32 in-place (final conv layer) --------------------
template<int F>
__global__ void self_bias_f32_k(float* __restrict__ agg, const float* __restrict__ h,
                                const float* __restrict__ isd, const float* __restrict__ bias,
                                long long N) {
    long long t = gtid();
    if (t >= N * F) return;
    long long n = t / F;
    int f = (int)(t % F);
    float is = isd[n];
    agg[t] = agg[t] + h[t] * is * is + bias[f];
}

// ---------------- pooling (float4) ----------------
__global__ void pool_sum_k(const float* __restrict__ z, const long long* __restrict__ batch,
                           float* __restrict__ sums, long long N) {
    long long t = gtid();
    if (t >= N * 16) return;
    long long n = t >> 4;
    int f4 = (int)(t & 15);
    long long g = batch[n];
    float4 v = ((const float4*)(z + n * 64))[f4];
    float* sp = sums + g * 64 + (size_t)f4 * 4;
    atomicAdd(sp + 0, v.x);
    atomicAdd(sp + 1, v.y);
    atomicAdd(sp + 2, v.z);
    atomicAdd(sp + 3, v.w);
}

__global__ void count_k(const long long* __restrict__ batch, float* __restrict__ cnt, long long N) {
    long long t = gtid();
    if (t < N) atomicAdd(&cnt[batch[t]], 1.0f);
}

// ---------------- fc + sigmoid ----------------
__global__ void final_k(const float* __restrict__ sums, const float* __restrict__ cnt,
                        const float* __restrict__ Wfc, const float* __restrict__ bfc,
                        float* __restrict__ out, long long G) {
    long long g = gtid();
    if (g >= G) return;
    float c = fmaxf(cnt[g], 1.0f);
    float acc = bfc[0];
#pragma unroll
    for (int f = 0; f < 64; ++f)
        acc += (sums[g * 64 + f] / c) * Wfc[f];
    out[g] = 1.0f / (1.0f + expf(-acc));
}

static inline unsigned nblocks(long long n) {
    return (unsigned)((n + THREADS - 1) / THREADS);
}

extern "C" void kernel_launch(void* const* d_in, const int* in_sizes, int n_in,
                              void* d_out, int out_size, void* d_ws, size_t ws_size,
                              hipStream_t stream) {
    const float*     x     = (const float*)d_in[0];
    const long long* ei    = (const long long*)d_in[1];
    const long long* batch = (const long long*)d_in[2];
    const float*     W1    = (const float*)d_in[3];
    const float*     b1    = (const float*)d_in[4];
    const float*     W2    = (const float*)d_in[5];
    const float*     b2    = (const float*)d_in[6];
    const float*     W3    = (const float*)d_in[7];
    const float*     b3    = (const float*)d_in[8];
    const float*     Wfc   = (const float*)d_in[9];
    const float*     bfc   = (const float*)d_in[10];
    float*           out   = (float*)d_out;

    const long long N = in_sizes[0];            // 100000
    const long long E = in_sizes[1] / 2;        // 3200000
    const long long G = out_size;               // 4096

    const long long* src = ei;
    const long long* dst = ei + E;

    // workspace carve-up
    float* ws    = (float*)d_ws;
    size_t NB    = (size_t)N * 128;
    float* bufA  = ws;                          // h (pre-aggregation), f32
    float* bufB  = bufA + NB;                   // agg buffer, f32
    float* deg   = bufB + NB;
    float* isd   = deg + N;
    float* sums  = isd + N;
    float* cnt   = sums + (size_t)G * 64;
    _Float16* act16 = (_Float16*)(cnt + G);     // f16 activations [N,128 max]
    _Float16* Wt2   = act16 + NB;               // W2^T f16 [128][64]
    _Float16* Wt3   = Wt2 + 64 * 128;           // W3^T f16 [64][128]

    const int fullTiles = (int)(N / 16);        // N=100000 -> 6250, no tail
    const int tailStart = fullTiles * 16;
    const int tailRows  = (int)(N - tailStart);
    const unsigned gemmGrid = (unsigned)((fullTiles + 7) / 8);  // 8 waves / block

    // weight transposition/conversion (tiny)
    wt_f16_k<<<nblocks(64 * 128), THREADS, 0, stream>>>(W2, Wt2, 64, 128);
    wt_f16_k<<<nblocks(128 * 64), THREADS, 0, stream>>>(W3, Wt3, 128, 64);

    // degrees + normalization
    zero_f<<<nblocks(N), THREADS, 0, stream>>>(deg, N);
    degree_k<<<nblocks(E), THREADS, 0, stream>>>(dst, deg, E);
    invsqrt_k<<<nblocks(N), THREADS, 0, stream>>>(deg, isd, N);

    // ---- layer 1: 1 -> 64, relu ----
    layer1_k<<<nblocks(N * 64), THREADS, 0, stream>>>(x, W1, bufA, N);
    zero_f<<<nblocks(N * 64), THREADS, 0, stream>>>(bufB, N * 64);
    aggregate_k<64><<<nblocks(E * 16), THREADS, 0, stream>>>(bufA, src, dst, isd, bufB, E);
    self_bias_relu_f16_k<64><<<nblocks(N * 64), THREADS, 0, stream>>>(bufB, bufA, isd, b1, act16, N);

    // ---- layer 2: 64 -> 128 (WMMA), relu ----
    gemm_wmma_k<64, 128><<<gemmGrid, THREADS, 0, stream>>>(act16, Wt2, bufA, fullTiles);
    if (tailRows > 0)
        gemm_tail_k<64, 128><<<nblocks((long long)tailRows * 128), THREADS, 0, stream>>>(
            act16, Wt2, bufA, tailStart, (int)N);
    zero_f<<<nblocks(N * 128), THREADS, 0, stream>>>(bufB, N * 128);
    aggregate_k<128><<<nblocks(E * 32), THREADS, 0, stream>>>(bufA, src, dst, isd, bufB, E);
    self_bias_relu_f16_k<128><<<nblocks(N * 128), THREADS, 0, stream>>>(bufB, bufA, isd, b2, act16, N);

    // ---- layer 3: 128 -> 64 (WMMA), no relu ----
    gemm_wmma_k<128, 64><<<gemmGrid, THREADS, 0, stream>>>(act16, Wt3, bufA, fullTiles);
    if (tailRows > 0)
        gemm_tail_k<128, 64><<<nblocks((long long)tailRows * 64), THREADS, 0, stream>>>(
            act16, Wt3, bufA, tailStart, (int)N);
    zero_f<<<nblocks(N * 64), THREADS, 0, stream>>>(bufB, N * 64);
    aggregate_k<64><<<nblocks(E * 16), THREADS, 0, stream>>>(bufA, src, dst, isd, bufB, E);
    self_bias_f32_k<64><<<nblocks(N * 64), THREADS, 0, stream>>>(bufB, bufA, isd, b3, N);

    // ---- global mean pool + fc + sigmoid ----
    zero_f<<<nblocks(G * 64), THREADS, 0, stream>>>(sums, G * 64);
    zero_f<<<nblocks(G), THREADS, 0, stream>>>(cnt, G);
    pool_sum_k<<<nblocks(N * 16), THREADS, 0, stream>>>(bufB, batch, sums, N);
    count_k<<<nblocks(N), THREADS, 0, stream>>>(batch, cnt, N);
    final_k<<<nblocks(G), THREADS, 0, stream>>>(sums, cnt, Wfc, bfc, out, G);
}
